// Generator_14654428414183
// MI455X (gfx1250) — matrix operand order repaired and checked by reading, not compiled, not run on hardware
//
#include <hip/hip_runtime.h>
#include <hip/hip_bf16.h>
#include <stdint.h>

typedef __attribute__((ext_vector_type(16))) __bf16 v16bf;
typedef __attribute__((ext_vector_type(8)))  float  v8f;

#define NJ 27
// Joint neighbor bitmasks: bit ji set in NBR[jo] <=> input joint ji feeds output joint jo.
// Derived by hand-executing get_neighbor(PARENTS, CONTACTS, 2, enforce_contact=True),
// including the contact-list aliasing and the root (joint 26) wiring.
__constant__ unsigned NBR[NJ] = {
  0x04000667u, 0x0400022Fu, 0x0400001Fu, 0x0040001Eu, 0x0080001Cu,
  0x040002E3u, 0x040001E1u, 0x010001E0u, 0x020001C0u, 0x04000E23u,
  0x04045E01u, 0x000CFE00u, 0x00047C00u, 0x00003800u, 0x0005DC00u,
  0x0003C800u, 0x0003C000u, 0x00038000u, 0x001C5C00u, 0x003C0800u,
  0x003C0000u, 0x00380000u, 0x0440001Eu, 0x0480001Cu, 0x050001E0u,
  0x060001C0u, 0x07C00667u
};

__device__ __forceinline__ unsigned short f2bf(float f) {
  unsigned u = __float_as_uint(f);
  u += 0x7FFFu + ((u >> 16) & 1u);     // round-to-nearest-even
  return (unsigned short)(u >> 16);
}

union BFrag { v16bf v; uint4 q[2]; };

// ---------------------------------------------------------------------------
// Weight prep: mask by joint adjacency, convert to bf16, pre-swizzle into the
// exact 16x32 A-fragment lane layout (ISA 7.12.2, 16-bit A matrix):
//   lane l holds row M = l&15; element i (VGPR v=i/2, half p=i&1):
//   K = (v<4 ? 2v+p : 16+2(v-4)+p) + (l>=16 ? 8 : 0)
// Fragment order [mtile][ctile][k] (k innermost, 1024 B per fragment) so the
// conv inner k-loop walks contiguous memory with immediate offsets.
// Rows co >= Cout are zeroed so padded output channels compute to exactly 0.
// ---------------------------------------------------------------------------
__global__ void prep_weights(const float* __restrict__ W, unsigned short* __restrict__ out,
                             int Cin, int Cout, int Ctiles, int Mtiles,
                             int cpj_in, int cpj_out) {
  int idx = blockIdx.x * blockDim.x + threadIdx.x;
  int total = 15 * Mtiles * Ctiles * 32;
  if (idx >= total) return;
  int lane = idx & 31;
  int r = idx >> 5;
  int k  = r % 15; r /= 15;
  int ct = r % Ctiles;
  int mt = r / Ctiles;
  int co  = mt * 16 + (lane & 15);
  int hi8 = (lane & 16) ? 8 : 0;
  __align__(16) unsigned short vals[16];
#pragma unroll
  for (int i = 0; i < 16; ++i) {
    int v = i >> 1, p = i & 1;
    int kl = (v < 4) ? (2 * v + p) : (16 + 2 * (v - 4) + p);
    int ci = ct * 32 + kl + hi8;
    float w = 0.f;
    if (co < Cout && ci < Cin) {
      int jo = co / cpj_out, ji = ci / cpj_in;
      if ((NBR[jo] >> ji) & 1u) w = W[((size_t)co * Cin + ci) * 15 + k];
    }
    vals[i] = f2bf(w);
  }
  unsigned short* dst = out + ((size_t)(((mt * Ctiles + ct) * 15 + k) * 32 + lane)) * 16;
  ((uint4*)dst)[0] = ((const uint4*)vals)[0];
  ((uint4*)dst)[1] = ((const uint4*)vals)[1];
}

// ---------------------------------------------------------------------------
// Stage-0 input: x0[b][t][c] = bf16(generated + noise0), channel-padded to Cp.
// ---------------------------------------------------------------------------
__global__ void make_x0(const float* __restrict__ gen, const float* __restrict__ nz,
                        unsigned short* __restrict__ out, int T, int C, int Cp) {
  size_t idx = (size_t)blockIdx.x * blockDim.x + threadIdx.x;
  size_t total = (size_t)8 * T * Cp;
  if (idx >= total) return;
  int c = (int)(idx % Cp);
  size_t r = idx / Cp;
  int t = (int)(r % T);
  int b = (int)(r / T);
  unsigned short o = 0;
  if (c < C) {
    size_t off = ((size_t)b * C + c) * T + t;
    o = f2bf(gen[off] + nz[off]);
  }
  out[idx] = o;
}

// ---------------------------------------------------------------------------
// Linear interpolate g0 (T0 -> T1, align_corners=False), keep gi in f32 for the
// final residual, and build stage-1 bf16 input x1 = bf16(gi + noise1), padded.
// ---------------------------------------------------------------------------
__global__ void interp_x1(const float* __restrict__ g0, const float* __restrict__ nz,
                          float* __restrict__ gi, unsigned short* __restrict__ x1,
                          int T0, int T1, int C, int Cp) {
  size_t idx = (size_t)blockIdx.x * blockDim.x + threadIdx.x;
  size_t total = (size_t)8 * T1 * Cp;
  if (idx >= total) return;
  int c = (int)(idx % Cp);
  size_t r = idx / Cp;
  int t = (int)(r % T1);
  int b = (int)(r / T1);
  unsigned short o = 0;
  if (c < C) {
    float src = (t + 0.5f) * ((float)T0 / (float)T1) - 0.5f;
    src = fminf(fmaxf(src, 0.f), (float)(T0 - 1));
    int i0 = (int)src;
    int i1 = min(i0 + 1, T0 - 1);
    float w = src - (float)i0;
    size_t base = ((size_t)b * C + c) * T0;
    float g = g0[base + i0] * (1.f - w) + g0[base + i1] * w;
    size_t od = ((size_t)b * C + c) * T1 + t;
    gi[od] = g;
    o = f2bf(g + nz[od]);
  }
  x1[idx] = o;
}

// Epilogue for one 16x16 output tile held in a WMMA f32 accumulator.
// D layout: lane holds column n = lane&15; VGPR v -> co = co0 + v.
__device__ __forceinline__ void conv_epilogue(
    v8f acc, int b, int T, int t, int co0, int Cout,
    const float* __restrict__ bias,
    unsigned short* __restrict__ Yb, int Coutp,
    float* __restrict__ Yf, const float* __restrict__ Res, int final_mode) {
  if (!final_mode) {
    unsigned pk[4];
#pragma unroll
    for (int vv = 0; vv < 4; ++vv) {
      int c0 = co0 + 2 * vv, c1 = c0 + 1;
      float x0 = acc[2 * vv]     + ((c0 < Cout) ? bias[c0] : 0.f);
      float x1 = acc[2 * vv + 1] + ((c1 < Cout) ? bias[c1] : 0.f);
      x0 = (x0 > 0.f) ? x0 : 0.2f * x0;
      x1 = (x1 > 0.f) ? x1 : 0.2f * x1;
      pk[vv] = (unsigned)f2bf(x0) | ((unsigned)f2bf(x1) << 16);
    }
    *(uint4*)(Yb + ((size_t)(b * T + t)) * Coutp + co0) =
        make_uint4(pk[0], pk[1], pk[2], pk[3]);
  } else {
#pragma unroll
    for (int vv = 0; vv < 8; ++vv) {
      int co = co0 + vv;
      if (co < Cout) {
        size_t off = ((size_t)b * Cout + co) * T + t;
        Yf[off] = acc[vv] + bias[co] + Res[off];
      }
    }
  }
}

// ---------------------------------------------------------------------------
// Skeleton conv layer via bf16 WMMA. Workgroup: LDS tile X[78][Cinp] (bf16,
// reflect-padded halo of 7) filled with GLOBAL_LOAD_ASYNC_TO_LDS_B128.
// 8 waves = 4 co-tiles x 2 t-halves; each wave keeps TWO independent 16x16
// accumulators (t and t+16) sharing each A fragment -> half the A traffic and
// two interleavable WMMA chains (hides D->C accumulate latency, ISA 7.12.1).
// All-zero (joint-masked) ci chunks are skipped wave-uniformly via NBR masks.
// ---------------------------------------------------------------------------
__global__ __launch_bounds__(256) void skel_conv(
    const unsigned short* __restrict__ X, int T, int Cinp, int Cin, int Ctiles, int cpj_in,
    const unsigned short* __restrict__ Wf, const float* __restrict__ bias,
    int Cout, int Mtiles, int cpj_out,
    unsigned short* __restrict__ Yb, int Coutp,
    float* __restrict__ Yf, const float* __restrict__ Res, int final_mode) {
  extern __shared__ __align__(16) unsigned short s_x[];
  const int b  = blockIdx.z;
  const int t0 = blockIdx.x * 64;
  const int tid = threadIdx.x;

  // Async cooperative LDS fill: rows t0-7 .. t0+70 (reflect at edges).
  // Per-lane memory->LDS b128 copies, tracked by ASYNCcnt (no VGPR staging).
  const int CW = Cinp >> 3;
  const int tot = 78 * CW;
  for (int i = tid; i < tot; i += 256) {
    int r = i / CW, c = i - r * CW;
    int t = t0 - 7 + r;
    if (t < 0) t = -t;
    else if (t >= T) t = 2 * T - 2 - t;
    const unsigned short* src = X + ((size_t)(b * T + t)) * Cinp + c * 8;
    unsigned ldsa = (unsigned)(uintptr_t)(s_x + (size_t)r * Cinp + c * 8); // addr[31:0] = LDS offset
    asm volatile("global_load_async_to_lds_b128 %0, %1, off"
                 :: "v"(ldsa), "v"(src) : "memory");
  }
  asm volatile("s_wait_asynccnt 0x0" ::: "memory");
  __syncthreads();

  const int wave = tid >> 5, lane = tid & 31;
  const int mt   = blockIdx.y * 4 + (wave >> 1);
  const int tsub = wave & 1;                 // each wave covers 32 timesteps
  if (mt >= Mtiles) return;

  // Union of output-joint neighbor masks for this co-tile (wave-uniform).
  unsigned nbru = 0;
  {
    int co_lo = mt * 16;
    if (co_lo < Cout) {
      int co_hi = min(co_lo + 15, Cout - 1);
      int jhi = min(co_hi / cpj_out, NJ - 1);
      for (int jo = co_lo / cpj_out; jo <= jhi; ++jo) nbru |= NBR[jo];
    }
  }

  const int n   = lane & 15;
  const int hiK = (lane & 16) ? 8 : 0;    // B: lane=column N, elems 0-7 K=hiK+s, elems 8-15 K=16+hiK+s-8
  v8f acc0 = {0.f, 0.f, 0.f, 0.f, 0.f, 0.f, 0.f, 0.f};
  v8f acc1 = acc0;

  for (int ct = 0; ct < Ctiles; ++ct) {
    int ci_lo = ct * 32;
    if (ci_lo >= Cin) break;
    int ci_hi = min(ci_lo + 31, Cin - 1);
    int jlo = ci_lo / cpj_in;
    int jhi = min(ci_hi / cpj_in, NJ - 1);
    unsigned cm = ((1u << (jhi - jlo + 1)) - 1u) << jlo;
    if (!(nbru & cm)) continue;          // masked-out block: skip 30 WMMAs
    // Per-(mt,ct) fragment block: 15 contiguous 1024-B A fragments (k-major).
    const unsigned short* wct = Wf +
        ((size_t)((mt * Ctiles + ct) * 15) * 32 + lane) * 16;
    const unsigned short* lbase =
        s_x + (size_t)(tsub * 32 + n) * Cinp + ci_lo + hiK;
    for (int k = 0; k < 15; ++k) {
      // A fragment: 32 B contiguous per lane at constant stride 1024 B per k
      // -> folds into the load's immediate offset; shared by both accs.
      const uint4* ap = (const uint4*)(wct + (size_t)k * 512);
      BFrag a;
      a.q[0] = ap[0];
      a.q[1] = ap[1];
      // B fragments: LDS row r = tsub*32 + (sub*16) + n + k
      const unsigned short* lrow0 = lbase + (size_t)k * Cinp;
      const unsigned short* lrow1 = lrow0 + (size_t)16 * Cinp;
      BFrag b0, b1;
      b0.q[0] = *(const uint4*)(lrow0);
      b0.q[1] = *(const uint4*)(lrow0 + 16);
      b1.q[0] = *(const uint4*)(lrow1);
      b1.q[1] = *(const uint4*)(lrow1 + 16);
      acc0 = __builtin_amdgcn_wmma_f32_16x16x32_bf16(
          false, a.v, false, b0.v, (short)0, acc0, false, false);
      acc1 = __builtin_amdgcn_wmma_f32_16x16x32_bf16(
          false, a.v, false, b1.v, (short)0, acc1, false, false);
    }
  }

  const int co0 = mt * 16 + ((lane & 16) ? 8 : 0);
  const int t_a = t0 + tsub * 32 + n;
  conv_epilogue(acc0, b, T, t_a,      co0, Cout, bias, Yb, Coutp, Yf, Res, final_mode);
  conv_epilogue(acc1, b, T, t_a + 16, co0, Cout, bias, Yb, Coutp, Yf, Res, final_mode);
}

// ---------------------------------------------------------------------------
// Host side
// ---------------------------------------------------------------------------
struct LayerCfg { int Cinp, Cin, Ctiles, Cout, Mtiles, Coutp, cpj_in, cpj_out, final_mode; };

// CH = [162,135,270,270,162]; pads: roundup32(Cin), Mtiles = Coutp/16 (hidden) or ceil(162/16)
static const LayerCfg LCFG[4] = {
  {192, 162, 6, 135, 10, 160,  6,  5, 0},
  {160, 135, 5, 270, 18, 288,  5, 10, 0},
  {288, 270, 9, 270, 18, 288, 10, 10, 0},
  {288, 270, 9, 162, 11,   0, 10,  6, 1},
};

extern "C" void kernel_launch(void* const* d_in, const int* in_sizes, int n_in,
                              void* d_out, int out_size, void* d_ws, size_t ws_size,
                              hipStream_t stream) {
  (void)in_sizes; (void)n_in; (void)out_size; (void)ws_size;
  const float* noise0 = (const float*)d_in[0];
  const float* gener  = (const float*)d_in[1];
  const float* noise1 = (const float*)d_in[2];

  const int B = 8, C = 162, T0 = 4096, T1 = 8192;

  // Workspace layout
  char* ws = (char*)d_ws;
  const size_t ACT = (size_t)B * T1 * 288 * 2;                 // 37,748,736 B per buffer
  unsigned short* actA = (unsigned short*)ws;
  unsigned short* actB = (unsigned short*)(ws + ACT);
  float* gi = (float*)(ws + 2 * ACT);                          // [B][162][T1] f32
  char* wbase = ws + 2 * ACT + (size_t)B * C * T1 * 4;
  unsigned short* wbuf[2][4];
  {
    size_t off = 0;
    for (int s = 0; s < 2; ++s)
      for (int l = 0; l < 4; ++l) {
        wbuf[s][l] = (unsigned short*)(wbase + off);
        off += (size_t)15 * LCFG[l].Mtiles * LCFG[l].Ctiles * 1024;
      }
  }

  // 1) Mask + swizzle + bf16-convert all 8 weight tensors.
  for (int s = 0; s < 2; ++s)
    for (int l = 0; l < 4; ++l) {
      const LayerCfg& L = LCFG[l];
      const float* W = (const float*)d_in[3 + s * 8 + l * 2];
      int total = 15 * L.Mtiles * L.Ctiles * 32;
      prep_weights<<<(total + 255) / 256, 256, 0, stream>>>(
          W, wbuf[s][l], L.Cin, L.Cout, L.Ctiles, L.Mtiles, L.cpj_in, L.cpj_out);
    }

  float* g0 = (float*)d_out;                       // [8][162][4096]
  float* g1 = g0 + (size_t)B * C * T0;             // [8][162][8192]

  // 2) Stage 0
  {
    size_t total = (size_t)B * T0 * 192;
    make_x0<<<(unsigned)((total + 255) / 256), 256, 0, stream>>>(gener, noise0, actA, T0, C, 192);
  }
  const unsigned short* src = actA;
  unsigned short* dst = actB;
  for (int l = 0; l < 4; ++l) {
    const LayerCfg& L = LCFG[l];
    const float* bias = (const float*)d_in[3 + 0 * 8 + l * 2 + 1];
    dim3 grid(T0 / 64, (L.Mtiles + 3) / 4, B);
    size_t lds = (size_t)78 * L.Cinp * 2;
    skel_conv<<<grid, 256, lds, stream>>>(
        src, T0, L.Cinp, L.Cin, L.Ctiles, L.cpj_in,
        wbuf[0][l], bias, L.Cout, L.Mtiles, L.cpj_out,
        dst, L.Coutp, g0, gener, L.final_mode);
    const unsigned short* t = dst; dst = (unsigned short*)src; src = t;
  }

  // 3) Upsample + stage-1 input
  {
    size_t total = (size_t)B * T1 * 192;
    interp_x1<<<(unsigned)((total + 255) / 256), 256, 0, stream>>>(
        g0, noise1, gi, actA, T0, T1, C, 192);
  }

  // 4) Stage 1
  src = actA; dst = actB;
  for (int l = 0; l < 4; ++l) {
    const LayerCfg& L = LCFG[l];
    const float* bias = (const float*)d_in[3 + 1 * 8 + l * 2 + 1];
    dim3 grid(T1 / 64, (L.Mtiles + 3) / 4, B);
    size_t lds = (size_t)78 * L.Cinp * 2;
    skel_conv<<<grid, 256, lds, stream>>>(
        src, T1, L.Cinp, L.Cin, L.Ctiles, L.cpj_in,
        wbuf[1][l], bias, L.Cout, L.Mtiles, L.cpj_out,
        dst, L.Coutp, g1, gi, L.final_mode);
    const unsigned short* t = dst; dst = (unsigned short*)src; src = t;
  }
}
